// DAGLayer_68925635166811
// MI455X (gfx1250) — matrix operand order repaired and checked by reading, not compiled, not run on hardware
//
#include <hip/hip_runtime.h>
#include <hip/hip_bf16.h>
#include <cstdint>

// ---------------------------------------------------------------------------
// DAG attention-GRU layer for MI455X (gfx1250, wave32, WMMA).
//
// Roofline reasoning:
//  * Edge GEMM (E x 128 x 128, 34 GFLOP) is algebraically eliminated:
//      alpha = hprev[i].wa_q + h[j].wa_k + ea.(We^T wa_k) + (be.wa_k + ba)
//    so the edge phase is bandwidth bound: ~2 x 268 MB of edge_attr @ 23.3TB/s.
//  * GRU gate GEMMs (rows x 128 -> rows x 384) are the dense compute; they run
//    on v_wmma_f32_16x16x32_bf16 with f32 accumulation, one 16-row tile/wave,
//    over *compacted* touched-node lists per level (not all N every level).
//  * Segment softmax via encoded-uint atomicMax + f32 atomicAdd (den, Sv),
//    normalization folded into the node-update kernel (msg = Sv/den).
//  * GRU activations use hardware transcendentals (v_exp_f32 / v_rcp_f32)
//    instead of IEEE division / ocml tanh, so they co-execute with the WMMA
//    pipe instead of serializing the elementwise tail.
// ---------------------------------------------------------------------------

typedef __bf16 bf16;
typedef __attribute__((ext_vector_type(16))) __bf16 v16bf;
typedef __attribute__((ext_vector_type(8)))  float  v8f;

static constexpr int NN = 65536;          // nodes
static constexpr int DD = 128;            // feature dim
static constexpr int EE = 524288;         // edges
static constexpr int LL = 8;              // levels
static constexpr int EL = EE / LL;        // edges per level (65536)
static constexpr int D3 = 3 * DD;         // 384 (GRU gate width)

#define DEVINL __device__ __forceinline__

DEVINL v8f wmma_bf16(v16bf a, v16bf b, v8f c) {
  // (neg_a, A, neg_b, B, c_mod, C, reuse_a, reuse_b)
  return __builtin_amdgcn_wmma_f32_16x16x32_bf16(false, a, false, b, (short)0, c,
                                                 false, false);
}

// A fragment (16x32 bf16) from a row-major 16x128 bf16 LDS tile.
// Layout per ISA 7.12.2: lane = m + 16*g; VGPR v holds K = g*8 + 2v+h (v<4)
// and K = 16 + g*8 + 2(v-4)+h (v>=4).
DEVINL v16bf load_a_frag_lds(const bf16* tile, int lane, int ks) {
  v16bf a;
  int m = lane & 15;
  int g = lane >> 4;
#pragma unroll
  for (int t = 0; t < 16; ++t) {
    int v = t >> 1, h = t & 1;
    int kl = (v < 4) ? (g * 8 + v * 2 + h) : (16 + g * 8 + (v - 4) * 2 + h);
    a[t] = tile[m * DD + ks * 32 + kl];
  }
  return a;
}

// B fragment (32x16 bf16) = W^T tile: B[k, o] = W[o, k], W row-major [384,128].
// B rows (K) striped across lane groups: lane = n + 16*kg, element t -> k = kg*8 + t (t<8), 16 + kg*8 + (t-8) (t>=8).
DEVINL v16bf load_bT_frag(const bf16* __restrict__ W, int lane, int ocol0, int ks) {
  v16bf b;
  int o = ocol0 + (lane & 15);
  int kg = lane >> 4;
#pragma unroll
  for (int t = 0; t < 16; ++t) {
    b[t] = W[o * DD + ks * 32 + ((t < 8) ? (kg * 8 + t) : (16 + kg * 8 + (t - 8)))];
  }
  return b;
}

// sigmoid via v_exp_f32 + v_rcp_f32 (no IEEE division sequence)
DEVINL float sigm(float x) { return __builtin_amdgcn_rcpf(1.0f + __expf(-x)); }
// tanh(x) = 1 - 2/(exp(2x)+1), same hardware transcendental path
DEVINL float tanh_fast(float x) {
  return 1.0f - 2.0f * __builtin_amdgcn_rcpf(__expf(2.0f * x) + 1.0f);
}

// Monotone float <-> uint encoding for atomicMax on floats of either sign.
DEVINL unsigned fenc(float f) {
  unsigned b = __float_as_uint(f);
  return (b & 0x80000000u) ? ~b : (b | 0x80000000u);
}
DEVINL float fdec(unsigned e) {
  unsigned b = (e & 0x80000000u) ? (e & 0x7fffffffu) : ~e;
  return __uint_as_float(b);
}

// ---------------------------------------------------------------------------
// Utility kernels
// ---------------------------------------------------------------------------

__global__ void k_zero(unsigned* __restrict__ p, size_t nwords) {
  size_t i = (size_t)blockIdx.x * blockDim.x + threadIdx.x;
  size_t stride = (size_t)gridDim.x * blockDim.x;
  for (; i < nwords; i += stride) p[i] = 0u;
}

__global__ void k_convert_weights(const float* __restrict__ Wi,
                                  const float* __restrict__ Wh,
                                  bf16* __restrict__ Wib, bf16* __restrict__ Whb) {
  int i = blockIdx.x * blockDim.x + threadIdx.x;
  if (i < D3 * DD) {
    Wib[i] = (bf16)Wi[i];
    Whb[i] = (bf16)Wh[i];
  }
}

// u = We^T @ wa_k ;  c = be . wa_k + ba   (folds the edge GEMM into a dot)
__global__ void k_prep_uc(const float* __restrict__ We, const float* __restrict__ be,
                          const float* __restrict__ wa, const float* __restrict__ ba,
                          float* __restrict__ u, float* __restrict__ cbuf) {
  int d = threadIdx.x;  // 128 threads
  float s = 0.0f;
  for (int dp = 0; dp < DD; ++dp) s += wa[DD + dp] * We[dp * DD + d];
  u[d] = s;
  if (d == 0) {
    float cv = ba[0];
    for (int dp = 0; dp < DD; ++dp) cv += be[dp] * wa[DD + dp];
    cbuf[0] = cv;
  }
}

// flag[target[e]] = 1 over all E edges (root = never a target)
__global__ void k_mark(const int* __restrict__ tgt, int* __restrict__ flag, int n) {
  int i = blockIdx.x * blockDim.x + threadIdx.x;
  if (i < n) flag[tgt[i]] = 1;
}

__global__ void k_compact(const int* __restrict__ flag, int* __restrict__ cnt,
                          int* __restrict__ list) {
  int i = blockIdx.x * blockDim.x + threadIdx.x;
  if (i < NN && flag[i]) {
    int p = atomicAdd(cnt, 1);
    list[p] = i;
  }
}

__global__ void k_relu(const float* __restrict__ in, float* __restrict__ out, size_t n) {
  size_t i = (size_t)blockIdx.x * blockDim.x + threadIdx.x;
  size_t stride = (size_t)gridDim.x * blockDim.x;
  for (; i < n; i += stride) out[i] = fmaxf(in[i], 0.0f);
}

// ---------------------------------------------------------------------------
// Root init: h = root ? GRU(h_prev, 0) : 0.  gh = bh (msg == 0), gi via WMMA.
// One wave (32 lanes) per 16-node tile; block = 1 wave (uniform early flow,
// EXEC all-ones around the WMMAs).
// ---------------------------------------------------------------------------
__global__ __launch_bounds__(32) void k_root_init(
    const float* __restrict__ hprev, float* __restrict__ hout,
    const int* __restrict__ nroot, const bf16* __restrict__ Wib,
    const float* __restrict__ bi, const float* __restrict__ bh) {
  __shared__ bf16 hp_s[16 * DD];
  __shared__ int rootv[16];
  int tile = blockIdx.x;
  int lane = threadIdx.x;
  int row0 = tile * 16;
  if (lane < 16) rootv[lane] = (nroot[row0 + lane] == 0);
  for (int idx = lane; idx < 16 * DD; idx += 32)
    hp_s[idx] = (bf16)hprev[(size_t)row0 * DD + idx];
  __syncthreads();

  v16bf ah[4];
#pragma unroll
  for (int ks = 0; ks < 4; ++ks) ah[ks] = load_a_frag_lds(hp_s, lane, ks);

  int ncol = lane & 15, gq = lane >> 4;
  for (int ot = 0; ot < 8; ++ot) {
    int d = ot * 16 + ncol;
    v8f cr, cz, cn;
    float br = bi[d], bz = bi[DD + d], bn = bi[2 * DD + d];
#pragma unroll
    for (int q = 0; q < 8; ++q) { cr[q] = br; cz[q] = bz; cn[q] = bn; }
#pragma unroll
    for (int ks = 0; ks < 4; ++ks) {
      cr = wmma_bf16(ah[ks], load_bT_frag(Wib, lane, ot * 16, ks), cr);
      cz = wmma_bf16(ah[ks], load_bT_frag(Wib, lane, DD + ot * 16, ks), cz);
      cn = wmma_bf16(ah[ks], load_bT_frag(Wib, lane, 2 * DD + ot * 16, ks), cn);
    }
    float ghr = bh[d], ghz = bh[DD + d], ghn = bh[2 * DD + d];
#pragma unroll
    for (int q = 0; q < 8; ++q) {
      int mrow = q + gq * 8;                // C-layout: VGPR q, lane group gq
      int node = row0 + mrow;
      float r = sigm(cr[q] + ghr);
      float z = sigm(cz[q] + ghz);
      float nv = tanh_fast(cn[q] + r * ghn);
      float hp = hprev[(size_t)node * DD + d];
      float val = rootv[mrow] ? ((1.0f - z) * nv + z * hp) : 0.0f;
      hout[(size_t)node * DD + d] = val;
    }
  }
}

// ---------------------------------------------------------------------------
// Per-edge attention scalar + segment max.  One wave per edge; 3 length-128
// dots via float4 loads + wave shuffle reduction.
// ---------------------------------------------------------------------------
__global__ __launch_bounds__(256) void k_alpha(
    const float* __restrict__ hprev, const float* __restrict__ h,
    const float* __restrict__ ea, const int* __restrict__ jsrc,
    const int* __restrict__ idst, const float* __restrict__ wa,
    const float* __restrict__ u, const float* __restrict__ cbuf,
    float* __restrict__ alpha, unsigned* __restrict__ menc,
    int* __restrict__ flag) {
  int wid = (int)((blockIdx.x * blockDim.x + threadIdx.x) >> 5);
  int lane = threadIdx.x & 31;
  if (wid >= EL) return;
  int j = jsrc[wid], i = idst[wid];
  int base = lane * 4;
  const float4 q  = *reinterpret_cast<const float4*>(hprev + (size_t)i * DD + base);
  const float4 wq = *reinterpret_cast<const float4*>(wa + base);
  const float4 k  = *reinterpret_cast<const float4*>(h + (size_t)j * DD + base);
  const float4 wk = *reinterpret_cast<const float4*>(wa + DD + base);
  const float4 e4 = *reinterpret_cast<const float4*>(ea + (size_t)wid * DD + base);
  const float4 u4 = *reinterpret_cast<const float4*>(u + base);
  float s = q.x * wq.x + q.y * wq.y + q.z * wq.z + q.w * wq.w +
            k.x * wk.x + k.y * wk.y + k.z * wk.z + k.w * wk.w +
            e4.x * u4.x + e4.y * u4.y + e4.z * u4.z + e4.w * u4.w;
#pragma unroll
  for (int off = 16; off > 0; off >>= 1) s += __shfl_xor(s, off, 32);
  if (lane == 0) {
    s += cbuf[0];
    alpha[wid] = s;
    atomicMax(menc + i, fenc(s));
    flag[i] = 1;
  }
}

// exp(alpha - max) -> den atomic, and unnormalized Sv += ex * h[j]
__global__ __launch_bounds__(256) void k_expsum(
    const float* __restrict__ h, const int* __restrict__ jsrc,
    const int* __restrict__ idst, const float* __restrict__ alpha,
    const unsigned* __restrict__ menc, float* __restrict__ den,
    float* __restrict__ Sv) {
  int wid = (int)((blockIdx.x * blockDim.x + threadIdx.x) >> 5);
  int lane = threadIdx.x & 31;
  if (wid >= EL) return;
  int j = jsrc[wid], i = idst[wid];
  float ex = __expf(alpha[wid] - fdec(menc[i]));
  int base = lane * 4;
  const float4 v = *reinterpret_cast<const float4*>(h + (size_t)j * DD + base);
  float* dst = Sv + (size_t)i * DD + base;
  atomicAdd(dst + 0, ex * v.x);
  atomicAdd(dst + 1, ex * v.y);
  atomicAdd(dst + 2, ex * v.z);
  atomicAdd(dst + 3, ex * v.w);
  if (lane == 0) atomicAdd(den + i, ex);
}

// ---------------------------------------------------------------------------
// GRU update over the compacted touched-node list.  One wave / 16-node tile,
// 6 accumulators (gi_r/z/n, gh_r/z/n) per 16-col output tile, GRU elementwise
// directly in the WMMA C-register layout.
// ---------------------------------------------------------------------------
__global__ __launch_bounds__(32) void k_node_update(
    const float* __restrict__ hprev, float* __restrict__ h,
    const float* __restrict__ Sv, const float* __restrict__ den,
    const int* __restrict__ list, const int* __restrict__ cnt,
    const bf16* __restrict__ Wib, const bf16* __restrict__ Whb,
    const float* __restrict__ bi, const float* __restrict__ bh) {
  __shared__ int nodes[16];
  __shared__ int valid[16];
  __shared__ float invd[16];
  __shared__ bf16 hp_s[16 * DD];
  __shared__ bf16 mg_s[16 * DD];

  int c = *cnt;
  int tile = blockIdx.x;
  if (tile * 16 >= c) return;  // block == 1 wave: uniform exit
  int lane = threadIdx.x;
  if (lane < 16) {
    int idx = tile * 16 + lane;
    int cl = idx < c ? idx : (c - 1);     // clamp pad rows onto a valid node
    int nd = list[cl];
    nodes[lane] = nd;
    valid[lane] = (idx < c);
    float dn = den[nd];
    invd[lane] = dn > 0.0f ? (1.0f / dn) : 0.0f;
  }
  __syncthreads();
  for (int idx = lane; idx < 16 * DD; idx += 32) {
    int rw = idx >> 7, col = idx & (DD - 1);
    int nd = nodes[rw];
    hp_s[idx] = (bf16)hprev[(size_t)nd * DD + col];
    mg_s[idx] = (bf16)(Sv[(size_t)nd * DD + col] * invd[rw]);  // msg = Sv/den
  }
  __syncthreads();

  v16bf ah[4], am[4];
#pragma unroll
  for (int ks = 0; ks < 4; ++ks) {
    ah[ks] = load_a_frag_lds(hp_s, lane, ks);
    am[ks] = load_a_frag_lds(mg_s, lane, ks);
  }

  int ncol = lane & 15, gq = lane >> 4;
  for (int ot = 0; ot < 8; ++ot) {
    int d = ot * 16 + ncol;
    v8f ir, iz, in_, hr, hz, hn;
    float bir = bi[d], biz = bi[DD + d], bin = bi[2 * DD + d];
    float bhr = bh[d], bhz = bh[DD + d], bhn = bh[2 * DD + d];
#pragma unroll
    for (int q = 0; q < 8; ++q) {
      ir[q] = bir; iz[q] = biz; in_[q] = bin;
      hr[q] = bhr; hz[q] = bhz; hn[q] = bhn;
    }
#pragma unroll
    for (int ks = 0; ks < 4; ++ks) {
      ir  = wmma_bf16(ah[ks], load_bT_frag(Wib, lane, ot * 16, ks), ir);
      iz  = wmma_bf16(ah[ks], load_bT_frag(Wib, lane, DD + ot * 16, ks), iz);
      in_ = wmma_bf16(ah[ks], load_bT_frag(Wib, lane, 2 * DD + ot * 16, ks), in_);
      hr  = wmma_bf16(am[ks], load_bT_frag(Whb, lane, ot * 16, ks), hr);
      hz  = wmma_bf16(am[ks], load_bT_frag(Whb, lane, DD + ot * 16, ks), hz);
      hn  = wmma_bf16(am[ks], load_bT_frag(Whb, lane, 2 * DD + ot * 16, ks), hn);
    }
#pragma unroll
    for (int q = 0; q < 8; ++q) {
      int mrow = q + gq * 8;
      if (!valid[mrow]) continue;
      int nd = nodes[mrow];
      float r = sigm(ir[q] + hr[q]);
      float z = sigm(iz[q] + hz[q]);
      float nv = tanh_fast(in_[q] + r * hn[q]);
      float hp = hprev[(size_t)nd * DD + d];
      h[(size_t)nd * DD + d] = (1.0f - z) * nv + z * hp;
    }
  }
}

// ---------------------------------------------------------------------------
// Host orchestration
// ---------------------------------------------------------------------------

struct DirParams {
  const float *We, *be, *wa, *ba, *Wi, *Wh, *bi, *bh;
};

extern "C" void kernel_launch(void* const* d_in, const int* in_sizes, int n_in,
                              void* d_out, int out_size, void* d_ws, size_t ws_size,
                              hipStream_t stream) {
  (void)in_sizes; (void)n_in; (void)out_size; (void)ws_size;

  const float* x  = (const float*)d_in[0];
  const int*   ei = (const int*)d_in[1];
  const float* ea = (const float*)d_in[2];
  // d_in[3] = num_levels (hardcoded LL=8)
  DirParams pf{(const float*)d_in[4],  (const float*)d_in[5],
               (const float*)d_in[6],  (const float*)d_in[7],
               (const float*)d_in[8],  (const float*)d_in[9],
               (const float*)d_in[10], (const float*)d_in[11]};
  DirParams pb{(const float*)d_in[12], (const float*)d_in[13],
               (const float*)d_in[14], (const float*)d_in[15],
               (const float*)d_in[16], (const float*)d_in[17],
               (const float*)d_in[18], (const float*)d_in[19]};

  char* ws = (char*)d_ws;
  size_t off = 0;
  auto take = [&](size_t bytes) -> char* {
    char* p = ws + off;
    off = (off + bytes + 255) & ~(size_t)255;
    return p;
  };
  float*    hA    = (float*)take((size_t)NN * DD * 4);
  float*    hB    = (float*)take((size_t)NN * DD * 4);
  float*    Sv    = (float*)take((size_t)NN * DD * 4);
  bf16*     Wib   = (bf16*)take((size_t)D3 * DD * 2);
  bf16*     Whb   = (bf16*)take((size_t)D3 * DD * 2);
  float*    ubuf  = (float*)take(DD * 4);
  float*    cbuf  = (float*)take(256);
  unsigned* zone  = (unsigned*)take((size_t)3 * NN * 4 + 256);  // m|den|flag|cnt
  unsigned* menc  = zone;
  float*    den   = (float*)(zone + NN);
  int*      flag  = (int*)(zone + 2 * NN);
  int*      cnt   = (int*)(zone + 3 * NN);
  int*      list  = (int*)take((size_t)NN * 4);
  int*      nroot = (int*)take((size_t)NN * 4);
  float*    alph  = (float*)take((size_t)EL * 4);

  const int* src_all = ei;       // edge_index[0]
  const int* dst_all = ei + EE;  // edge_index[1]

  auto run_dir = [&](const float* hprev, float* hout, const DirParams& P,
                     const int* jall, const int* iall, bool rev) {
    k_convert_weights<<<(D3 * DD + 255) / 256, 256, 0, stream>>>(P.Wi, P.Wh, Wib, Whb);
    k_prep_uc<<<1, DD, 0, stream>>>(P.We, P.be, P.wa, P.ba, ubuf, cbuf);
    k_zero<<<512, 256, 0, stream>>>((unsigned*)nroot, (size_t)NN);
    k_mark<<<EE / 256, 256, 0, stream>>>(iall, nroot, EE);
    k_root_init<<<NN / 16, 32, 0, stream>>>(hprev, hout, nroot, Wib, P.bi, P.bh);
    for (int l = 0; l < LL; ++l) {
      int lev = rev ? (LL - 1 - l) : l;
      const int* jl = jall + (size_t)lev * EL;
      const int* il = iall + (size_t)lev * EL;
      const float* eal = ea + (size_t)lev * EL * DD;
      k_zero<<<1024, 256, 0, stream>>>(zone, (size_t)3 * NN + 64);
      k_zero<<<2048, 256, 0, stream>>>((unsigned*)Sv, (size_t)NN * DD);
      k_alpha<<<EL / 8, 256, 0, stream>>>(hprev, hout, eal, jl, il, P.wa, ubuf,
                                          cbuf, alph, menc, flag);
      k_expsum<<<EL / 8, 256, 0, stream>>>(hout, jl, il, alph, menc, den, Sv);
      k_compact<<<NN / 256, 256, 0, stream>>>(flag, cnt, list);
      k_node_update<<<EL / 16, 32, 0, stream>>>(hprev, hout, Sv, den, list, cnt,
                                                Wib, Whb, P.bi, P.bh);
    }
  };

  // forward: aggregate at dst, h_prev = x
  run_dir(x, hA, pf, src_all, dst_all, false);
  // backward: aggregate at src, levels reversed, h_prev = h_fwd
  run_dir(hA, hB, pb, dst_all, src_all, true);

  k_relu<<<2048, 256, 0, stream>>>(hB, (float*)d_out, (size_t)NN * DD);
}
